// PhysicalModelNetwork_28595892257379
// MI455X (gfx1250) — matrix operand order repaired and checked by reading, not compiled, not run on hardware
//
#include <hip/hip_runtime.h>
#include <hip/hip_bf16.h>

// ============================================================================
// PhysicalModelNetwork for MI455X (gfx1250, wave32, WMMA).
// All matmuls -> v_wmma_f32_16x16x32_bf16 (bf16 operands, f32 accum).
//  - rfft magnitude   == GEMM vs precomputed cos/sin DFT bases
//  - to_spec + irfft  == GEMM vs one precomputed fused (257x512) matrix
//  - recurrence runs entirely inside one workgroup per 16 batch rows
// ============================================================================

typedef __bf16 bf16;
typedef __attribute__((ext_vector_type(16))) __bf16 v16bf;
typedef __attribute__((ext_vector_type(8)))  __bf16 v8bf;
typedef __attribute__((ext_vector_type(8)))  float  v8f;

#define C_     257
#define CP     288          // padded to 18*16 (9 k-tiles of 32)
#define NFR    128
#define NS     32768
#define WIN    512
#define STEP_  256
#define BATCH_ 512
#define OSTR   512          // uniform f32 output-tile stride in LDS

struct StackW {
  const bf16*  w[4];   // WMMA-fragment-ordered bf16, (288x288), last = wo
  const float* b[4];   // padded biases (288)
  const float* g[3];   // layernorm gamma (288, pads=0)
  const float* e[3];   // layernorm beta  (288, pads=0)
};

// ---------------- WMMA fragment loaders (A in this code's contiguous k order; D per ISA) ----------------
// A (16x32 bf16, MxK), k order used here: lanes 0-15 => M=lane, K 0..15;
//                      lanes16-31 => M=lane-16, K 16..31.
// abase = Ab + m*astride + (lane>>4)*16 ; per-kt step = 32 elements (64 B).
__device__ __forceinline__ v16bf load_a_off(const bf16* abase, int koff) {
  v8bf lo = *(const v8bf*)(abase + koff);
  v8bf hi = *(const v8bf*)(abase + koff + 8);
  return __builtin_shufflevector(lo, hi, 0,1,2,3,4,5,6,7,8,9,10,11,12,13,14,15);
}
// D (16x16 f32): reg r => row (lane>>4)*8 + r, col lane&15.
__device__ __forceinline__ void store_d(float* Ob, int ostride, int nt, v8f acc) {
  int lane = threadIdx.x & 31;
  int n    = nt * 16 + (lane & 15);
  int mb   = (lane >> 4) * 8;
#pragma unroll
  for (int r = 0; r < 8; ++r) Ob[(mb + r) * ostride + n] = acc[r];
}

// 16-token x (KT*32) @ (KT*32 x NTt*16) GEMM. Block = 256 threads (8 waves);
// wave w owns n-tiles {w, w+8, w+16, w+24} that exist. Fully unrolled:
// fixed accumulators (no movrel), scalar-branch for the one partial tile.
template <int KT, int NTt>
__device__ __forceinline__ void gemm_tile(const bf16* __restrict__ Ab, int astride,
                                          const bf16* __restrict__ frag,
                                          const float* __restrict__ bias,
                                          float* __restrict__ Ob, int ostride) {
  constexpr int NPW = (NTt + 7) / 8;                 // n-tiles per wave (max)
  const int wave = __builtin_amdgcn_readfirstlane((int)threadIdx.x) >> 5;  // SGPR
  const int lane = threadIdx.x & 31;

  const bf16* abase = Ab + (lane & 15) * astride + (lane >> 4) * 16;
  const bf16* bbase[NPW];
  v8f acc[NPW];
#pragma unroll
  for (int i = 0; i < NPW; ++i) {
    const int nt = wave + 8 * i;
    bbase[i] = frag + (((size_t)nt * 32 + lane) << 4);
    float bv = 0.f;
    if ((8 * i + 7 < NTt) || (nt < NTt))
      bv = bias ? bias[nt * 16 + (lane & 15)] : 0.f;
#pragma unroll
    for (int r = 0; r < 8; ++r) acc[i][r] = bv;
  }

#pragma unroll
  for (int kt = 0; kt < KT; ++kt) {
    v16bf a = load_a_off(abase, kt * 32);
    if (kt + 1 < KT)   // gfx1250 global_prefetch_b8 on next weight k-tile
      __builtin_prefetch(bbase[0] + (size_t)(kt + 1) * NTt * 512, 0, 1);
#pragma unroll
    for (int i = 0; i < NPW; ++i) {
      if ((8 * i + 7 < NTt) || (wave + 8 * i < NTt)) {   // scalar cond (SGPR wave)
        v16bf b = *(const v16bf*)(bbase[i] + (size_t)kt * NTt * 512);
        acc[i] = __builtin_amdgcn_wmma_f32_16x16x32_bf16(
            false, a, false, b, (short)0, acc[i], false, false);
      }
    }
  }
#pragma unroll
  for (int i = 0; i < NPW; ++i)
    if ((8 * i + 7 < NTt) || (wave + 8 * i < NTt))
      store_d(Ob, ostride, wave + 8 * i, acc[i]);
}

// per-row (16 rows x 16 threads) reduction helper; red = float[16*32]
__device__ __forceinline__ float reduce_row16(float* red, float v) {
  int row = threadIdx.x >> 4, sub = threadIdx.x & 15;
  red[row * 32 + sub] = v;
  __syncthreads();
  float s = 0.f;
#pragma unroll
  for (int i = 0; i < 16; ++i) s += red[row * 32 + i];
  __syncthreads();
  return s;
}

// LayerNorm (over 257 real cols; pads are zero) + leaky_relu(0.2) -> bf16 Ab
__device__ __forceinline__ void ln_leaky(const float* Ob, const float* g, const float* e,
                                         bf16* Ab, float* red) {
  int row = threadIdx.x >> 4, sub = threadIdx.x & 15;
  float s = 0.f, s2 = 0.f;
  for (int c = sub; c < CP; c += 16) { float v = Ob[row * OSTR + c]; s += v; s2 += v * v; }
  red[row * 32 + sub] = s; red[row * 32 + 16 + sub] = s2;
  __syncthreads();
  float m = 0.f, q = 0.f;
#pragma unroll
  for (int i = 0; i < 16; ++i) { m += red[row * 32 + i]; q += red[row * 32 + 16 + i]; }
  m *= (1.f / 257.f); q = q * (1.f / 257.f) - m * m;
  float inv = rsqrtf(q + 1e-5f);
  __syncthreads();
  for (int c = sub; c < CP; c += 16) {
    float v = (Ob[row * OSTR + c] - m) * inv * g[c] + e[c];   // pads: g=e=0 -> 0
    v = v > 0.f ? v : 0.2f * v;
    Ab[row * CP + c] = (bf16)v;
  }
  __syncthreads();
}

// LinearOutputStack: 3x [Linear -> LN -> leaky] + final Linear. Result in Ob.
// Layer loop unrolled so S.w[l]/S.b[l] are direct SGPR args (no cndmask chains).
__device__ __forceinline__ void run_stack(const StackW& S, bf16* Ab, float* Ob, float* red) {
#pragma unroll
  for (int l = 0; l < 3; ++l) {
    gemm_tile<9, 18>(Ab, CP, S.w[l], S.b[l], Ob, OSTR);
    __syncthreads();
    ln_leaky(Ob, S.g[l], S.e[l], Ab, red);
  }
  gemm_tile<9, 18>(Ab, CP, S.w[3], S.b[3], Ob, OSTR);
  __syncthreads();
}

// ============================ kernels =======================================

// frames -> DFT magnitude (2 GEMMs) -> NPN(embed) -> control latents (bf16)
__global__ __launch_bounds__(256)
void embed_audio_kernel(const float* __restrict__ cs,
                        const bf16* __restrict__ cosF, const bf16* __restrict__ sinF,
                        StackW E, bf16* __restrict__ ctrl) {
  extern __shared__ char smem[];
  bf16*  Fb  = (bf16*)smem;                                   // 16x512 bf16
  bf16*  Ab  = (bf16*)(smem + 16 * WIN * 2);                  // 16x288 bf16
  float* Ob  = (float*)(smem + 16 * WIN * 2 + 16 * CP * 2);   // 16x512 f32
  float* O2  = Ob + 16 * OSTR;                                // 16x512 f32
  float* red = O2 + 16 * OSTR;                                // 16x32
  float* inn = red + 16 * 32;                                 // 16
  int row = threadIdx.x >> 4, sub = threadIdx.x & 15;
  int token = blockIdx.x * 16 + row;
  int bb = token >> 7, f = token & 127;
  const float TWO_PI = 6.28318530717958647692f;

  for (int j = sub; j < WIN; j += 16) {                       // Hann-windowed frame
    int pos = f * STEP_ + j;
    float w = 0.5f * (1.f - cosf(TWO_PI * (float)j / (float)WIN));
    float v = (pos < NS) ? cs[(size_t)bb * NS + pos] * w : 0.f;
    Fb[row * WIN + j] = (bf16)v;
  }
  __syncthreads();
  gemm_tile<16, 18>(Fb, WIN, cosF, nullptr, Ob, OSTR);  __syncthreads();
  gemm_tile<16, 18>(Fb, WIN, sinF, nullptr, O2, OSTR);  __syncthreads();

  float part = 0.f;                                           // |rfft| + input norm
  for (int c = sub; c < CP; c += 16) {
    float re = Ob[row * OSTR + c], im = O2[row * OSTR + c];
    part += re * re + im * im;
    Ab[row * CP + c] = (bf16)sqrtf(re * re + im * im);
  }
  float ss = reduce_row16(red, part);
  if (sub == 0) inn[row] = sqrtf(ss);
  __syncthreads();

  run_stack(E, Ab, Ob, red);                                  // NPN MLP

  float p2 = 0.f;                                             // relu + unit-norm * inn
  for (int c = sub; c < CP; c += 16) { float v = fmaxf(Ob[row * OSTR + c], 0.f); p2 += v * v; }
  float tn = reduce_row16(red, p2);
  float sc = inn[row] / (sqrtf(tn) + 1e-8f);
  for (int c = sub; c < CP; c += 16)
    ctrl[(size_t)token * CP + c] = (bf16)(fmaxf(Ob[row * OSTR + c], 0.f) * sc);
}

// shape -> NPN(embed_shape) -> shape latents (bf16)
__global__ __launch_bounds__(256)
void embed_shape_kernel(const float* __restrict__ shp, StackW E, bf16* __restrict__ slat) {
  extern __shared__ char smem[];
  bf16*  Ab  = (bf16*)smem;                                   // 16x288 bf16
  float* Ob  = (float*)(smem + 16 * CP * 2);                  // 16x512 f32
  float* red = Ob + 16 * OSTR;
  float* inn = red + 16 * 32;
  int row = threadIdx.x >> 4, sub = threadIdx.x & 15;
  int token = blockIdx.x * 16 + row;

  float part = 0.f;
  for (int c = sub; c < CP; c += 16) {
    float v = (c < C_) ? shp[(size_t)token * C_ + c] : 0.f;
    part += v * v;
    Ab[row * CP + c] = (bf16)v;
  }
  float ss = reduce_row16(red, part);
  if (sub == 0) inn[row] = sqrtf(ss);
  __syncthreads();

  run_stack(E, Ab, Ob, red);

  float p2 = 0.f;
  for (int c = sub; c < CP; c += 16) { float v = fmaxf(Ob[row * OSTR + c], 0.f); p2 += v * v; }
  float tn = reduce_row16(red, p2);
  float sc = inn[row] / (sqrtf(tn) + 1e-8f);
  for (int c = sub; c < CP; c += 16)
    slat[(size_t)token * CP + c] = (bf16)(fmaxf(Ob[row * OSTR + c], 0.f) * sc);
}

// 128-step leaky-integrator recurrence; each block owns 16 independent rows.
__global__ __launch_bounds__(256)
void recur_kernel(const bf16* __restrict__ ctrl, const bf16* __restrict__ slat,
                  StackW T, StackW L, const bf16* __restrict__ fusF,
                  float* __restrict__ out) {
  extern __shared__ char smem[];
  float* hb   = (float*)smem;                                 // 16x288 f32 state
  bf16*  Ab   = (bf16*)(hb + 16 * CP);                        // 16x288 bf16
  float* Ob   = (float*)((char*)Ab + 16 * CP * 2);            // 16x512 f32
  float* tail = Ob + 16 * OSTR;                               // 16x256 OLA tail
  float* red  = tail + 16 * 256;                              // 16x32
  float* inn  = red + 16 * 32;                                // 16
  float* win  = inn + 16;                                     // 512
  int row = threadIdx.x >> 4, sub = threadIdx.x & 15;
  int bb = blockIdx.x * 16 + row;
  const float TWO_PI = 6.28318530717958647692f;

  for (int c = sub; c < CP;  c += 16)  hb[row * CP + c] = 0.f;
  for (int j = sub; j < 256; j += 16)  tail[row * 256 + j] = 0.f;
  for (int j = threadIdx.x; j < WIN; j += 256)
    win[j] = 0.5f * (1.f - cosf(TWO_PI * (float)j / (float)WIN));
  __syncthreads();

  for (int t = 0; t < NFR; ++t) {
    size_t lb = ((size_t)bb * NFR + t) * CP;
    // ---- x = h + control; norm(x) ----
    float part = 0.f;
    for (int c = sub; c < CP; c += 16) {
      float x = hb[row * CP + c] + (float)ctrl[lb + c];
      part += x * x;
      Ab[row * CP + c] = (bf16)x;
    }
    float ss = reduce_row16(red, part);
    if (sub == 0) inn[row] = sqrtf(ss);
    __syncthreads();
    // ---- h = NPN_transform(x) ----
    run_stack(T, Ab, Ob, red);
    float p2 = 0.f;
    for (int c = sub; c < CP; c += 16) { float v = fmaxf(Ob[row * OSTR + c], 0.f); p2 += v * v; }
    float tn = reduce_row16(red, p2);
    float sc = inn[row] / (sqrtf(tn) + 1e-8f);
    for (int c = sub; c < CP; c += 16) {
      float v = fmaxf(Ob[row * OSTR + c], 0.f) * sc;
      hb[row * CP + c] = v;                                   // new h
      Ab[row * CP + c] = (bf16)(v + (float)slat[lb + c]);     // leakage input
    }
    __syncthreads();
    // ---- lr = sigmoid(stack_leakage(h+sl)) * 0.5; leaked; h -= leaked ----
    run_stack(L, Ab, Ob, red);
    for (int c = sub; c < CP; c += 16) {
      float lr = 0.5f / (1.f + expf(-Ob[row * OSTR + c]));
      float h  = hb[row * CP + c];
      float lk = h * lr;                                      // pads: h=0 -> 0
      hb[row * CP + c] = h - lk;
      Ab[row * CP + c] = (bf16)lk;
    }
    __syncthreads();
    // ---- samples = leaked @ Fused (to_spec∘irfft), window, overlap-add ----
    gemm_tile<9, 32>(Ab, CP, fusF, nullptr, Ob, OSTR);
    __syncthreads();
    for (int j = sub; j < 256; j += 16) {
      float s = tail[row * 256 + j] + Ob[row * OSTR + j] * win[j];
      out[(size_t)bb * NS + t * STEP_ + j] = s;               // finalized sample
      tail[row * 256 + j] = Ob[row * OSTR + 256 + j] * win[256 + j];
    }
    __syncthreads();
  }
}

// ======================= parameter preprocessing ============================
// Swizzle (257x257) linear weight W (out = x @ W^T) into WMMA B-fragment
// order for B[k][n] = W[n][k], padded to 288x288, bf16.
__global__ void prep_linear_frag_kernel(const float* __restrict__ W, bf16* __restrict__ dst) {
  int idx = blockIdx.x * 256 + threadIdx.x;   // 9*18*512 = 82944
  int j = idx & 15, lane = (idx >> 4) & 31, tile = idx >> 9;
  int nt = tile % 18, kt = tile / 18;
  int n = nt * 16 + (lane & 15);
  int k = kt * 32 + (lane >> 4) * 16 + j;
  float v = (k < C_ && n < C_) ? W[n * C_ + k] : 0.f;
  dst[idx] = (bf16)v;
}

// DFT bases (512 x 257), cos & sin, fragment order, K=512 (16 k-tiles).
__global__ void prep_dft_frag_kernel(bf16* __restrict__ dc, bf16* __restrict__ ds) {
  int idx = blockIdx.x * 256 + threadIdx.x;   // 16*18*512 = 147456
  int j = idx & 15, lane = (idx >> 4) & 31, tile = idx >> 9;
  int nt = tile % 18, kt = tile / 18;
  int n = nt * 16 + (lane & 15);
  int k = kt * 32 + (lane >> 4) * 16 + j;
  float c = 0.f, s = 0.f;
  if (n < C_) {
    int ph = (k * n) & 511;                   // exp(-i*2pi*k*n/512), period 512
    float th = 6.28318530717958647692f * (float)ph / 512.f;
    c = cosf(th); s = sinf(th);               // sign irrelevant for |.|
  }
  dc[idx] = (bf16)c; ds[idx] = (bf16)s;
}

// Fused[k][n] = sum_q to_spec[q][k]*irC(q,n) + to_spec[257+q][k]*irS(q,n)
// (to_spec^T composed with the real-irfft basis), fragment order, K=257->288.
__global__ void prep_fused_frag_kernel(const float* __restrict__ ts, bf16* __restrict__ dst) {
  int idx = blockIdx.x * 256 + threadIdx.x;   // 9*32*512 = 147456
  int j = idx & 15, lane = (idx >> 4) & 31, tile = idx >> 9;
  int nt = tile & 31, kt = tile >> 5;
  int n = nt * 16 + (lane & 15);              // 0..511 sample
  int k = kt * 32 + (lane >> 4) * 16 + j;     // 0..287 coeff
  float v = 0.f;
  if (k < C_) {
    for (int q = 0; q < C_; ++q) {
      float cq = (q == 0 || q == 256) ? (1.f / 512.f) : (2.f / 512.f);
      int ph = (q * n) & 511;
      float th = 6.28318530717958647692f * (float)ph / 512.f;
      v += ts[q * C_ + k] * (cq * cosf(th)) - ts[(C_ + q) * C_ + k] * (cq * sinf(th));
    }
  }
  dst[idx] = (bf16)v;
}

__global__ void prep_vec_kernel(const float* __restrict__ src, float* __restrict__ dst) {
  int i = threadIdx.x;                        // 288 threads
  dst[i] = (i < C_) ? src[i] : 0.f;
}

// ============================== launch ======================================
extern "C" void kernel_launch(void* const* d_in, const int* in_sizes, int n_in,
                              void* d_out, int out_size, void* d_ws, size_t ws_size,
                              hipStream_t stream) {
  (void)in_sizes; (void)n_in; (void)out_size; (void)ws_size;
  const float* cs  = (const float*)d_in[0];
  const float* shp = (const float*)d_in[1];
  // jax pytree flattening: dicts by sorted key ->
  //  2..15 embed{b0,b1,b2,bo,e0,e1,e2,g0,g1,g2,w0,w1,w2,wo}, 16..29 embed_shape,
  //  30..43 leakage, 44 to_spec, 45..58 transform
  const int bases[4] = {2, 16, 30, 45};       // embed, embed_shape, leakage, transform
  const int TS_IDX = 44;

  char* w = (char*)d_ws;
  auto alloc = [&](size_t bytes) -> void* {
    void* p = (void*)w; w += (bytes + 255) & ~(size_t)255; return p;
  };
  const size_t FRAG_LIN = (size_t)9 * 18 * 512;   // elems
  const size_t FRAG_DFT = (size_t)16 * 18 * 512;
  const size_t FRAG_OUT = (size_t)9 * 32 * 512;

  StackW S[4];
  for (int s = 0; s < 4; ++s) {
    for (int i = 0; i < 4; ++i) S[s].w[i] = (bf16*)alloc(FRAG_LIN * 2);
    for (int i = 0; i < 4; ++i) S[s].b[i] = (float*)alloc(CP * 4);
    for (int i = 0; i < 3; ++i) S[s].g[i] = (float*)alloc(CP * 4);
    for (int i = 0; i < 3; ++i) S[s].e[i] = (float*)alloc(CP * 4);
  }
  bf16* cosF  = (bf16*)alloc(FRAG_DFT * 2);
  bf16* sinF  = (bf16*)alloc(FRAG_DFT * 2);
  bf16* fusF  = (bf16*)alloc(FRAG_OUT * 2);
  bf16* ctrlL = (bf16*)alloc((size_t)BATCH_ * NFR * CP * 2);
  bf16* shpL  = (bf16*)alloc((size_t)BATCH_ * NFR * CP * 2);

  for (int s = 0; s < 4; ++s) {
    int base = bases[s];
    for (int i = 0; i < 4; ++i)   // w0,w1,w2,wo at +10..+13
      prep_linear_frag_kernel<<<324, 256, 0, stream>>>(
          (const float*)d_in[base + 10 + i], (bf16*)S[s].w[i]);
    for (int i = 0; i < 4; ++i)   // b0..b2,bo at +0..+3
      prep_vec_kernel<<<1, CP, 0, stream>>>((const float*)d_in[base + i], (float*)S[s].b[i]);
    for (int i = 0; i < 3; ++i)   // g0..g2 at +7..+9
      prep_vec_kernel<<<1, CP, 0, stream>>>((const float*)d_in[base + 7 + i], (float*)S[s].g[i]);
    for (int i = 0; i < 3; ++i)   // e0..e2 at +4..+6
      prep_vec_kernel<<<1, CP, 0, stream>>>((const float*)d_in[base + 4 + i], (float*)S[s].e[i]);
  }
  prep_dft_frag_kernel<<<576, 256, 0, stream>>>(cosF, sinF);
  prep_fused_frag_kernel<<<576, 256, 0, stream>>>((const float*)d_in[TS_IDX], fusF);

  size_t lds_embed = 16 * WIN * 2 + 16 * CP * 2 + 2 * (16 * OSTR * 4) + 16 * 32 * 4 + 16 * 4;
  size_t lds_shape = 16 * CP * 2 + 16 * OSTR * 4 + 16 * 32 * 4 + 16 * 4;
  size_t lds_recur = 16 * CP * 4 + 16 * CP * 2 + 16 * OSTR * 4 + 16 * 256 * 4
                   + 16 * 32 * 4 + 16 * 4 + WIN * 4;
  (void)hipFuncSetAttribute((const void*)embed_audio_kernel,
      hipFuncAttributeMaxDynamicSharedMemorySize, (int)lds_embed);
  (void)hipFuncSetAttribute((const void*)embed_shape_kernel,
      hipFuncAttributeMaxDynamicSharedMemorySize, (int)lds_shape);
  (void)hipFuncSetAttribute((const void*)recur_kernel,
      hipFuncAttributeMaxDynamicSharedMemorySize, (int)lds_recur);

  embed_audio_kernel<<<4096, 256, lds_embed, stream>>>(cs, cosF, sinF, S[0], ctrlL);
  embed_shape_kernel<<<4096, 256, lds_shape, stream>>>(shp, S[1], shpL);
  recur_kernel<<<32, 256, lds_recur, stream>>>(ctrlL, shpL, S[3], S[2], fusF, (float*)d_out);
}
